// SelfAttentionEncoder_326417515275
// MI455X (gfx1250) — matrix operand order repaired and checked by reading, not compiled, not run on hardware
//
#include <hip/hip_runtime.h>
#include <cstdint>
#include <cstddef>

// Problem constants (B,S,D from the reference)
#define B_ 4
#define S_ 4096
#define D_ 512

typedef __attribute__((ext_vector_type(16))) __bf16 v16bf;
typedef __attribute__((ext_vector_type(8)))  float  v8f;

// 128x128 block tile, K-chunk 32, 8 waves (wave32) of 2x4 16x16 WMMA tiles each
#define BM 128
#define BN 128
#define BK 32
#define LDSS 40   // LDS row stride in bf16 elems (80 B: 16B-aligned rows, bank-skewed)
#define ABUF (BM * LDSS)   // one A/B buffer in bf16 elems

union FragU { v16bf v; uint4 q[2]; };

// ---------------------------------------------------------------------------
// Async tile copy: 128 rows x 32 bf16, global -> LDS, no VGPR round trip.
// Uses gfx1250 GLOBAL_LOAD_ASYNC_TO_LDS_B128 (ASYNCcnt-tracked).
// LDS byte address = low 32 bits of the generic pointer (ISA 10.2 aperture map).
// ---------------------------------------------------------------------------
__device__ __forceinline__ void issue_tile_async(
    const __bf16* __restrict__ src, int ld, int row0, int k0,
    __bf16* dst, int tid)
{
#pragma unroll
  for (int it = 0; it < 2; ++it) {
    int i   = tid + it * 256;
    int r   = i >> 2;          // 0..127
    int seg = (i & 3) << 3;    // 0,8,16,24 (bf16 elems => 16B each)
    const __bf16* g = src + (size_t)(row0 + r) * ld + k0 + seg;
    unsigned lds = (unsigned)(uintptr_t)(dst + r * LDSS + seg);
    asm volatile("global_load_async_to_lds_b128 %0, %1, off"
                 :: "v"(lds), "v"(g) : "memory");
  }
}

__device__ __forceinline__ void wait_async_and_barrier() {
  asm volatile("s_wait_asynccnt 0x0" ::: "memory");
  __syncthreads();
}

// A fragment (16x32 bf16, MxK): lane half h: elems 0..7 <-> K=h*8.., 8..15 <-> K=16+h*8..
__device__ __forceinline__ v16bf frag_a(const __bf16* As, int row, int half) {
  FragU f;
  const __bf16* base = As + row * LDSS;
  f.q[0] = *reinterpret_cast<const uint4*>(base + half * 8);
  f.q[1] = *reinterpret_cast<const uint4*>(base + 16 + half * 8);
  return f.v;
}

// B fragment (32x16 bf16, KxN): lane half h: elems 0..7 <-> K=h*8.., 8..15 <-> K=16+h*8.., N = lane&15
__device__ __forceinline__ v16bf frag_b(const __bf16* Bs, int ncol, int half) {
  FragU f;
  const __bf16* base = Bs + ncol * LDSS + half * 8;
  f.q[0] = *reinterpret_cast<const uint4*>(base);
  f.q[1] = *reinterpret_cast<const uint4*>(base + 16);
  return f.v;
}

// ---------------------------------------------------------------------------
// Shared double-buffered WMMA pipeline: acc[2][4] += A[m0.., :] @ Bm[n0.., :]^T
// A and Bm both row-major with leading dim ld (= K depth).
// As/Bs point to 2*ABUF bf16 of LDS each.
// ---------------------------------------------------------------------------
__device__ __forceinline__ void wmma_pipeline(
    const __bf16* __restrict__ A, const __bf16* __restrict__ Bm,
    int ld, int m0, int n0, int Kd,
    __bf16* As, __bf16* Bs,
    int tid, int half, int lrow, int mw, int nw,
    v8f acc[2][4])
{
  // prologue: stage chunk 0 into buffer 0
  issue_tile_async(A,  ld, m0, 0, As, tid);
  issue_tile_async(Bm, ld, n0, 0, Bs, tid);
  wait_async_and_barrier();

  const int nchunks = Kd / BK;
  for (int c = 0; c < nchunks; ++c) {
    __bf16* Ac = As + (c & 1) * ABUF;
    __bf16* Bc = Bs + (c & 1) * ABUF;
    // overlap: async-copy chunk c+1 into the other buffer while computing c
    if (c + 1 < nchunks) {
      issue_tile_async(A,  ld, m0, (c + 1) * BK, As + ((c + 1) & 1) * ABUF, tid);
      issue_tile_async(Bm, ld, n0, (c + 1) * BK, Bs + ((c + 1) & 1) * ABUF, tid);
    }
    v16bf bfr[4];
#pragma unroll
    for (int nt = 0; nt < 4; ++nt) bfr[nt] = frag_b(Bc, nw + nt * 16 + lrow, half);
#pragma unroll
    for (int mt = 0; mt < 2; ++mt) {
      v16bf afr = frag_a(Ac, mw + mt * 16 + lrow, half);
#pragma unroll
      for (int nt = 0; nt < 4; ++nt)
        acc[mt][nt] = __builtin_amdgcn_wmma_f32_16x16x32_bf16(
            false, afr, false, bfr[nt], (short)0, acc[mt][nt], false, false);
    }
    wait_async_and_barrier();
  }
}

// ---------------------------------------------------------------------------
// GEMM: out[MxN] = A[MxK] @ W[NxK]^T (+ bias[N]); out stored bf16 OR f32
// ---------------------------------------------------------------------------
__global__ __launch_bounds__(256) void gemm_bf16_kernel(
    const __bf16* __restrict__ A, const __bf16* __restrict__ W,
    const float* __restrict__ bias,
    __bf16* __restrict__ outb, float* __restrict__ outf,
    int M, int N, int Kd)
{
  __shared__ __bf16 As[2 * ABUF];
  __shared__ __bf16 Bs[2 * ABUF];

  const int tid  = threadIdx.x;
  const int lane = tid & 31, wave = tid >> 5;
  const int half = lane >> 4, lrow = lane & 15;
  const int wm = wave >> 1, wn = wave & 1;          // 4 (M) x 2 (N) waves
  const int m0 = blockIdx.x * BM, n0 = blockIdx.y * BN;
  const int mw = wm * 32, nw = wn * 64;

  v8f acc[2][4];
#pragma unroll
  for (int mt = 0; mt < 2; ++mt)
#pragma unroll
    for (int nt = 0; nt < 4; ++nt)
#pragma unroll
      for (int r = 0; r < 8; ++r) acc[mt][nt][r] = 0.0f;

  wmma_pipeline(A, W, Kd, m0, n0, Kd, As, Bs, tid, half, lrow, mw, nw, acc);

#pragma unroll
  for (int nt = 0; nt < 4; ++nt) {
    const int ncol = n0 + nw + nt * 16 + lrow;
    const float bb = bias ? bias[ncol] : 0.0f;
#pragma unroll
    for (int mt = 0; mt < 2; ++mt) {
#pragma unroll
      for (int r = 0; r < 8; ++r) {
        const int row = m0 + mw + mt * 16 + half * 8 + r;
        const float v = acc[mt][nt][r] + bb;
        if (outb) outb[(size_t)row * N + ncol] = (__bf16)v;
        else      outf[(size_t)row * N + ncol] = v;
      }
    }
  }
}

// ---------------------------------------------------------------------------
// Pass 1: colden[b,e] += sum_s exp(K[b,s,:].Q[b,e,:] / D)
// ---------------------------------------------------------------------------
__global__ __launch_bounds__(256) void qk_colsum_kernel(
    const __bf16* __restrict__ Kb, const __bf16* __restrict__ Qb,
    float* __restrict__ denom)
{
  __shared__ __bf16 As[2 * ABUF];
  __shared__ __bf16 Bs[2 * ABUF];
  __shared__ float colred[BN];

  const int tid  = threadIdx.x;
  const int lane = tid & 31, wave = tid >> 5;
  const int half = lane >> 4, lrow = lane & 15;
  const int wm = wave >> 1, wn = wave & 1;
  const int m0 = blockIdx.x * BM, n0 = blockIdx.y * BN;
  const int mw = wm * 32, nw = wn * 64;
  const int b  = blockIdx.z;

  const __bf16* A  = Kb + (size_t)b * S_ * D_;
  const __bf16* Bm = Qb + (size_t)b * S_ * D_;

  v8f acc[2][4];
#pragma unroll
  for (int mt = 0; mt < 2; ++mt)
#pragma unroll
    for (int nt = 0; nt < 4; ++nt)
#pragma unroll
      for (int r = 0; r < 8; ++r) acc[mt][nt][r] = 0.0f;

  wmma_pipeline(A, Bm, D_, m0, n0, D_, As, Bs, tid, half, lrow, mw, nw, acc);

  const float scale = 1.0f / (float)D_;
  float colp[4] = {0.f, 0.f, 0.f, 0.f};
#pragma unroll
  for (int nt = 0; nt < 4; ++nt)
#pragma unroll
    for (int mt = 0; mt < 2; ++mt)
#pragma unroll
      for (int r = 0; r < 8; ++r)
        colp[nt] += __expf(acc[mt][nt][r] * scale);
#pragma unroll
  for (int nt = 0; nt < 4; ++nt)
    colp[nt] += __shfl_xor(colp[nt], 16, 32);   // merge row halves (same column)

  for (int i = tid; i < BN; i += 256) colred[i] = 0.0f;
  __syncthreads();
  if (half == 0) {
#pragma unroll
    for (int nt = 0; nt < 4; ++nt)
      atomicAdd(&colred[nw + nt * 16 + lrow], colp[nt]);
  }
  __syncthreads();
  for (int i = tid; i < BN; i += 256)
    atomicAdd(&denom[(size_t)b * S_ + n0 + i], colred[i]);
}

// ---------------------------------------------------------------------------
// Pass 2: rows[b,s] += sum_e exp(K.Q / D) / colden[b,e]
// ---------------------------------------------------------------------------
__global__ __launch_bounds__(256) void qk_rowsum_kernel(
    const __bf16* __restrict__ Kb, const __bf16* __restrict__ Qb,
    const float* __restrict__ denom, float* __restrict__ rows)
{
  __shared__ __bf16 As[2 * ABUF];
  __shared__ __bf16 Bs[2 * ABUF];
  __shared__ float rowred[BM];

  const int tid  = threadIdx.x;
  const int lane = tid & 31, wave = tid >> 5;
  const int half = lane >> 4, lrow = lane & 15;
  const int wm = wave >> 1, wn = wave & 1;
  const int m0 = blockIdx.x * BM, n0 = blockIdx.y * BN;
  const int mw = wm * 32, nw = wn * 64;
  const int b  = blockIdx.z;

  const __bf16* A  = Kb + (size_t)b * S_ * D_;
  const __bf16* Bm = Qb + (size_t)b * S_ * D_;

  float invd[4];
#pragma unroll
  for (int nt = 0; nt < 4; ++nt)
    invd[nt] = 1.0f / denom[(size_t)b * S_ + n0 + nw + nt * 16 + lrow];

  v8f acc[2][4];
#pragma unroll
  for (int mt = 0; mt < 2; ++mt)
#pragma unroll
    for (int nt = 0; nt < 4; ++nt)
#pragma unroll
      for (int r = 0; r < 8; ++r) acc[mt][nt][r] = 0.0f;

  wmma_pipeline(A, Bm, D_, m0, n0, D_, As, Bs, tid, half, lrow, mw, nw, acc);

  const float scale = 1.0f / (float)D_;
  float rowp[2][8];
#pragma unroll
  for (int mt = 0; mt < 2; ++mt)
#pragma unroll
    for (int r = 0; r < 8; ++r) rowp[mt][r] = 0.0f;
#pragma unroll
  for (int mt = 0; mt < 2; ++mt)
#pragma unroll
    for (int nt = 0; nt < 4; ++nt)
#pragma unroll
      for (int r = 0; r < 8; ++r)
        rowp[mt][r] += __expf(acc[mt][nt][r] * scale) * invd[nt];

  // reduce across the 16 lanes of each half (same rows, different columns)
#pragma unroll
  for (int mt = 0; mt < 2; ++mt)
#pragma unroll
    for (int r = 0; r < 8; ++r) {
      float v = rowp[mt][r];
      v += __shfl_xor(v, 1, 32);
      v += __shfl_xor(v, 2, 32);
      v += __shfl_xor(v, 4, 32);
      v += __shfl_xor(v, 8, 32);
      rowp[mt][r] = v;
    }

  for (int i = tid; i < BM; i += 256) rowred[i] = 0.0f;
  __syncthreads();
  if (lrow == 0) {
#pragma unroll
    for (int mt = 0; mt < 2; ++mt)
#pragma unroll
      for (int r = 0; r < 8; ++r)
        atomicAdd(&rowred[mw + mt * 16 + half * 8 + r], rowp[mt][r]);
  }
  __syncthreads();
  for (int i = tid; i < BM; i += 256)
    atomicAdd(&rows[(size_t)b * S_ + m0 + i], rowred[i]);
}

// ---------------------------------------------------------------------------
// Small helper kernels
// ---------------------------------------------------------------------------
__global__ void f32_to_bf16_kernel(const float* __restrict__ src,
                                   __bf16* __restrict__ dst, int n) {
  int i = blockIdx.x * blockDim.x + threadIdx.x;
  if (i < n) dst[i] = (__bf16)src[i];
}

__global__ void zero_kernel(float* __restrict__ p, int n) {
  int i = blockIdx.x * blockDim.x + threadIdx.x;
  if (i < n) p[i] = 0.0f;
}

// Wvl = Wl @ Wv  (so xs @ Wvl^T == (xs @ Wv^T) @ Wl^T), stored bf16 [n][d]
__global__ void compose_wvl_kernel(const float* __restrict__ Wl,
                                   const float* __restrict__ Wv,
                                   __bf16* __restrict__ Wvlb) {
  int d = blockIdx.x * 16 + threadIdx.x;
  int n = blockIdx.y * 16 + threadIdx.y;
  float acc = 0.0f;
  for (int t = 0; t < D_; ++t) acc += Wl[n * D_ + t] * Wv[t * D_ + d];
  Wvlb[n * D_ + d] = (__bf16)acc;
}

// bvl[n] = sum_d Wl[n,d] * bv[d]
__global__ void compose_bvl_kernel(const float* __restrict__ Wl,
                                   const float* __restrict__ bv,
                                   float* __restrict__ bvl) {
  int n = blockIdx.x * blockDim.x + threadIdx.x;
  if (n < D_) {
    float a = 0.0f;
    for (int d = 0; d < D_; ++d) a += Wl[n * D_ + d] * bv[d];
    bvl[n] = a;
  }
}

// out = tanh(rows[b,s] * Vl[b,s,n] + bl[n])
__global__ void final_kernel(const float* __restrict__ Vl,
                             const float* __restrict__ rows,
                             const float* __restrict__ bl,
                             float* __restrict__ out) {
  size_t i = (size_t)blockIdx.x * blockDim.x + threadIdx.x;
  const size_t total = (size_t)B_ * S_ * D_;
  if (i < total) {
    size_t row = i / D_;
    int n = (int)(i % D_);
    out[i] = tanhf(rows[row] * Vl[i] + bl[n]);
  }
}

// ---------------------------------------------------------------------------
// Launcher
// ---------------------------------------------------------------------------
extern "C" void kernel_launch(void* const* d_in, const int* in_sizes, int n_in,
                              void* d_out, int out_size, void* d_ws, size_t ws_size,
                              hipStream_t stream) {
  const float* xs = (const float*)d_in[0];
  const float* Wk = (const float*)d_in[1];
  const float* bk = (const float*)d_in[2];
  const float* Wq = (const float*)d_in[3];
  const float* bq = (const float*)d_in[4];
  const float* Wv = (const float*)d_in[5];
  const float* bv = (const float*)d_in[6];
  const float* Wl = (const float*)d_in[7];
  const float* bl = (const float*)d_in[8];
  float* out = (float*)d_out;

  char* ws = (char*)d_ws;
  size_t o = 0;
  __bf16* xs_bf = (__bf16*)(ws + o); o += (size_t)B_ * S_ * D_ * 2;   // 16 MB
  __bf16* Kbf   = (__bf16*)(ws + o); o += (size_t)B_ * S_ * D_ * 2;   // 16 MB
  __bf16* Qbf   = (__bf16*)(ws + o); o += (size_t)B_ * S_ * D_ * 2;   // 16 MB
  float*  Vl    = (float*) (ws + o); o += (size_t)B_ * S_ * D_ * 4;   // 32 MB
  __bf16* Wkb   = (__bf16*)(ws + o); o += (size_t)D_ * D_ * 2;
  __bf16* Wqb   = (__bf16*)(ws + o); o += (size_t)D_ * D_ * 2;
  __bf16* Wvlb  = (__bf16*)(ws + o); o += (size_t)D_ * D_ * 2;
  float*  bvl   = (float*) (ws + o); o += 2048;
  float*  denom = (float*) (ws + o); o += (size_t)B_ * S_ * 4;        // followed by rows
  float*  rows  = (float*) (ws + o); o += (size_t)B_ * S_ * 4;

  const int M  = B_ * S_;       // 16384 rows for all projections
  const int nx = B_ * S_ * D_;  // 8388608 elements

  // 1) precision staging: xs / Wk / Wq -> bf16 ; fold Wl@Wv -> Wvl(bf16), bvl
  f32_to_bf16_kernel<<<dim3((nx + 255) / 256), dim3(256), 0, stream>>>(xs, xs_bf, nx);
  f32_to_bf16_kernel<<<dim3((D_ * D_ + 255) / 256), dim3(256), 0, stream>>>(Wk, Wkb, D_ * D_);
  f32_to_bf16_kernel<<<dim3((D_ * D_ + 255) / 256), dim3(256), 0, stream>>>(Wq, Wqb, D_ * D_);
  compose_wvl_kernel<<<dim3(D_ / 16, D_ / 16), dim3(16, 16), 0, stream>>>(Wl, Wv, Wvlb);
  compose_bvl_kernel<<<dim3(2), dim3(256), 0, stream>>>(Wl, bv, bvl);

  // 2) zero accumulation vectors (denom and rows are contiguous)
  zero_kernel<<<dim3((2 * B_ * S_ + 255) / 256), dim3(256), 0, stream>>>(denom, 2 * B_ * S_);

  // 3) projections via WMMA: K, Q (bf16 out) and fused Vl = xs@(Wl@Wv)^T + bvl (f32)
  dim3 ggrid(M / BM, D_ / BN);
  gemm_bf16_kernel<<<ggrid, dim3(256), 0, stream>>>(xs_bf, Wkb,  bk,  Kbf, nullptr, M, D_, D_);
  gemm_bf16_kernel<<<ggrid, dim3(256), 0, stream>>>(xs_bf, Wqb,  bq,  Qbf, nullptr, M, D_, D_);
  gemm_bf16_kernel<<<ggrid, dim3(256), 0, stream>>>(xs_bf, Wvlb, bvl, nullptr, Vl,  M, D_, D_);

  // 4) column-softmax denominators, then row-sums (two WMMA passes over QK^T)
  dim3 qkgrid(S_ / BM, S_ / BN, B_);
  qk_colsum_kernel<<<qkgrid, dim3(256), 0, stream>>>(Kbf, Qbf, denom);
  qk_rowsum_kernel<<<qkgrid, dim3(256), 0, stream>>>(Kbf, Qbf, denom, rows);

  // 5) out = tanh(rows * Vl + bl)
  final_kernel<<<dim3((nx + 255) / 256), dim3(256), 0, stream>>>(Vl, rows, bl, out);
}
